// GCN_57629871178505
// MI455X (gfx1250) — compile-verified, matrix operand order checked by reading, not run on hardware
//
#include <hip/hip_runtime.h>
#include <hip/hip_bf16.h>

// ---------------------------------------------------------------------------
// GATv2 x3 forward for MI455X (gfx1250, wave32, WMMA).
// Dense projections use v_wmma_f32_16x16x32_f16; edge passes use f32 atomics
// (node tables are L2-resident: 25.6MB each << 192MB L2).
// ---------------------------------------------------------------------------

#define IN_DIM   39
#define HID      64
#define OUT_DIM  2
#define EDGE_DIM 16
#define SLOPE    0.2f
#define KEEP_P   0.8f

typedef __attribute__((ext_vector_type(16))) _Float16 v16h;
typedef __attribute__((ext_vector_type(8)))  _Float16 v8h;
typedef __attribute__((ext_vector_type(8)))  float    v8f;

// ---- helpers --------------------------------------------------------------

__device__ __forceinline__ unsigned enc_f32(float f) {
    unsigned u = __float_as_uint(f);
    return (u & 0x80000000u) ? ~u : (u | 0x80000000u);   // order-preserving
}
__device__ __forceinline__ float dec_f32(unsigned u) {
    return (u & 0x80000000u) ? __uint_as_float(u ^ 0x80000000u)
                             : __uint_as_float(~u);
}
#define ENC_NEG_INF 0x007FFFFFu   // enc_f32(-inf)

__device__ __forceinline__ unsigned hash_u32(unsigned x) {
    x ^= x >> 16; x *= 0x7feb352du;
    x ^= x >> 15; x *= 0x846ca68bu;
    x ^= x >> 16; return x;
}

// ---- fill kernels ---------------------------------------------------------

__global__ void fill_f32_kernel(float* __restrict__ p, float v, long long n) {
    for (long long i = (long long)blockIdx.x * blockDim.x + threadIdx.x; i < n;
         i += (long long)gridDim.x * blockDim.x) p[i] = v;
}
__global__ void fill_u32_kernel(unsigned* __restrict__ p, unsigned v, long long n) {
    for (long long i = (long long)blockIdx.x * blockDim.x + threadIdx.x; i < n;
         i += (long long)gridDim.x * blockDim.x) p[i] = v;
}

// ---- self-loop attr: loop_attr[d] = mean of incoming edge_attr ------------

__global__ void loop_accum_kernel(const int* __restrict__ dst,
                                  const float* __restrict__ eattr,
                                  float* __restrict__ loop_attr,
                                  float* __restrict__ cnt, long long E16) {
    for (long long i = (long long)blockIdx.x * blockDim.x + threadIdx.x; i < E16;
         i += (long long)gridDim.x * blockDim.x) {
        long long e = i >> 4; int k = (int)(i & 15);
        int d = dst[e];
        atomicAdd(loop_attr + (size_t)d * EDGE_DIM + k, eattr[i]);
        if (k == 0) atomicAdd(cnt + d, 1.0f);
    }
}
__global__ void loop_div_kernel(float* __restrict__ loop_attr,
                                const float* __restrict__ cnt, long long N16) {
    for (long long i = (long long)blockIdx.x * blockDim.x + threadIdx.x; i < N16;
         i += (long long)gridDim.x * blockDim.x) {
        float c = cnt[i >> 4];
        loop_attr[i] /= fmaxf(c, 1.0f);
    }
}

// ---- f32 -> f16 feature conversion (zero-padded to 64 cols) ---------------

__global__ void cvt_x_kernel(const float* __restrict__ x, int K,
                             _Float16* __restrict__ Xh, long long N64) {
    for (long long i = (long long)blockIdx.x * blockDim.x + threadIdx.x; i < N64;
         i += (long long)gridDim.x * blockDim.x) {
        long long n = i >> 6; int j = (int)(i & 63);
        Xh[i] = (j < K) ? (_Float16)x[n * K + j] : (_Float16)0.0f;
    }
}

// ---- weight prep: W[K x D] f32 -> Wt[64 cols x 64 k] f16, zero-padded.
//      Also pads bias to 64 f32 entries (drops all guards from the GEMM). ----

__global__ void prep_wt_kernel(const float* __restrict__ W, int K, int D,
                               const float* __restrict__ bias,
                               _Float16* __restrict__ Wt,
                               float* __restrict__ biasp) {
    int idx = blockIdx.x * blockDim.x + threadIdx.x;   // 0..4095
    if (idx >= 64 * 64) return;
    int n = idx >> 6, k = idx & 63;
    Wt[n * 64 + k] = (k < K && n < D) ? (_Float16)W[k * D + n] : (_Float16)0.0f;
    if (idx < 64) biasp[idx] = (idx < D) ? bias[idx] : 0.0f;
}

// ---- WMMA GEMM: Y[N x 64] = Xh[N x 64] @ W (via Wt) + biasp ---------------
// One 16x16 tile per wave; 8 waves/block -> 32 rows x 64 cols per block.
// Fragment layout per ISA 7.12.2 (16-bit A 16x32): lanes 0-15 hold K 0..7 /
// 16..23, lanes 16-31 hold K 8..15 / 24..31. B mirrored with columns striped.

__global__ void gemm_wmma_kernel(const _Float16* __restrict__ Xh,
                                 const _Float16* __restrict__ Wt,
                                 const float* __restrict__ biasp,
                                 float* __restrict__ Y, int nrows) {
    const int wave = threadIdx.x >> 5;          // 0..7
    const int lane = threadIdx.x & 31;
    const int hl = lane & 15, hi = lane >> 4;
    const int row0 = blockIdx.x * 32 + (wave >> 2) * 16;
    const int col0 = (wave & 3) * 16;

    int m = row0 + hl; if (m >= nrows) m = nrows - 1;   // clamp for loads
    const int n = col0 + hl;

    const _Float16* xrow = Xh + (size_t)m * 64;
    const _Float16* wcol = Wt + (size_t)n * 64;

    v8f c = {};
#pragma unroll
    for (int kb = 0; kb < 64; kb += 32) {
        v8h alo = *(const v8h*)(xrow + kb + hi * 8);
        v8h ahi = *(const v8h*)(xrow + kb + 16 + hi * 8);
        v8h blo = *(const v8h*)(wcol + kb + hi * 8);
        v8h bhi = *(const v8h*)(wcol + kb + 16 + hi * 8);
        v16h a, b;
#pragma unroll
        for (int t = 0; t < 8; ++t) {
            a[t] = alo[t]; a[t + 8] = ahi[t];
            b[t] = blo[t]; b[t + 8] = bhi[t];
        }
        c = __builtin_amdgcn_wmma_f32_16x16x32_f16(
                /*neg_a=*/false, a, /*neg_b=*/false, b,
                /*c_mod=*/(short)0, c, /*reuse_a=*/false, /*reuse_b=*/false);
    }

    const float bn = biasp[n];
    // D layout: vgpr r holds element (m = hi*8 + r, n = hl).
    if (row0 + 16 <= nrows) {                   // wave-uniform fast path
        float* yp = Y + (size_t)(row0 + hi * 8) * 64 + n;
#pragma unroll
        for (int r = 0; r < 8; ++r) yp[(size_t)r * 64] = c[r] + bn;
    } else {
#pragma unroll
        for (int r = 0; r < 8; ++r) {
            int mm = row0 + hi * 8 + r;
            if (mm < nrows) Y[(size_t)mm * 64 + n] = c[r] + bn;
        }
    }
}

// ---- edge pass A: logits + encoded atomic segment-max ---------------------

__global__ void edge_logits_kernel(const int* __restrict__ src,
                                   const int* __restrict__ dst,
                                   const float* __restrict__ eattr,
                                   const float* __restrict__ loop_attr,
                                   const float* __restrict__ xl,
                                   const float* __restrict__ xr,
                                   const float* __restrict__ we,
                                   const float* __restrict__ att, int D,
                                   long long E, long long Ef,
                                   float* __restrict__ logits,
                                   unsigned* __restrict__ mxe) {
    __shared__ float s_we[EDGE_DIM * 64];
    __shared__ float s_att[64];
    for (int idx = threadIdx.x; idx < EDGE_DIM * 64; idx += blockDim.x) {
        int k = idx >> 6, j = idx & 63;
        s_we[idx] = (j < D) ? we[k * D + j] : 0.0f;
    }
    for (int j = threadIdx.x; j < 64; j += blockDim.x)
        s_att[j] = (j < D) ? att[j] : 0.0f;
    __syncthreads();

    const long long stride = (long long)gridDim.x * blockDim.x;
    for (long long i = (long long)blockIdx.x * blockDim.x + threadIdx.x; i < Ef;
         i += stride) {
        int s, d; const float* ep;
        if (i < E) { s = src[i]; d = dst[i]; ep = eattr + (size_t)i * EDGE_DIM; }
        else       { s = d = (int)(i - E);  ep = loop_attr + (size_t)s * EDGE_DIM; }

        // prefetch next edge_attr row of the HBM stream (global_prefetch_b8)
        if (i + stride < E)
            __builtin_prefetch(eattr + (size_t)(i + stride) * EDGE_DIM, 0, 3);

        float e[EDGE_DIM];
        const float4* ep4 = (const float4*)ep;
#pragma unroll
        for (int q = 0; q < 4; ++q) {
            float4 v = ep4[q];
            e[4 * q + 0] = v.x; e[4 * q + 1] = v.y;
            e[4 * q + 2] = v.z; e[4 * q + 3] = v.w;
        }
        const float* xls = xl + (size_t)s * 64;
        const float* xrd = xr + (size_t)d * 64;

        float acc = 0.0f;
        for (int j = 0; j < D; ++j) {
            float ee = 0.0f;
#pragma unroll
            for (int k = 0; k < EDGE_DIM; ++k) ee = fmaf(e[k], s_we[k * 64 + j], ee);
            float mv = xls[j] + xrd[j] + ee;
            mv = (mv > 0.0f) ? mv : SLOPE * mv;        // leaky relu
            acc = fmaf(s_att[j], mv, acc);
        }
        logits[i] = acc;
        atomicMax(mxe + d, enc_f32(acc));
    }
}

// ---- edge pass B: ex = exp(logit - mx[dst]); den[dst] += ex ---------------

__global__ void edge_exp_kernel(const int* __restrict__ dst,
                                const float* __restrict__ logits,
                                const unsigned* __restrict__ mxe,
                                float* __restrict__ den,
                                float* __restrict__ exv,
                                long long E, long long Ef) {
    for (long long i = (long long)blockIdx.x * blockDim.x + threadIdx.x; i < Ef;
         i += (long long)gridDim.x * blockDim.x) {
        int d = (i < E) ? dst[i] : (int)(i - E);
        float ex = __expf(logits[i] - dec_f32(mxe[d]));
        exv[i] = ex;
        atomicAdd(den + d, ex);
    }
}

// ---- edge pass C: out[dst] += alpha * xl[src]  (wave-per-edge, coalesced) --

__global__ void edge_scatter_kernel(const int* __restrict__ src,
                                    const int* __restrict__ dst,
                                    const float* __restrict__ exv,
                                    const float* __restrict__ den,
                                    const float* __restrict__ xl,
                                    float* __restrict__ outacc, int D,
                                    long long E, long long Ef) {
    long long gid = (long long)blockIdx.x * blockDim.x + threadIdx.x;
    long long ei = gid >> 5;
    int lane = (int)(gid & 31);
    if (ei >= Ef) return;

    int s, d;
    if (ei < E) { s = src[ei]; d = dst[ei]; }
    else        { s = d = (int)(ei - E); }

    float alpha = exv[ei] / den[d];                   // den>0: self-loop exists
    const float* xls = xl + (size_t)s * 64;
    float* od = outacc + (size_t)d * 64;
    for (int j = lane; j < D; j += 32)
        atomicAdd(od + j, alpha * xls[j]);
}

// ---- finalize: bias + relu + deterministic dropout -> f16 next features ----

__global__ void finalize_hidden_kernel(const float* __restrict__ outacc,
                                       const float* __restrict__ bias,
                                       unsigned seed,
                                       _Float16* __restrict__ Xh, long long N64) {
    const unsigned keep_thresh = (unsigned)(KEEP_P * 16777216.0f);
    for (long long i = (long long)blockIdx.x * blockDim.x + threadIdx.x; i < N64;
         i += (long long)gridDim.x * blockDim.x) {
        int j = (int)(i & 63);
        float v = outacc[i] + bias[j];
        v = fmaxf(v, 0.0f);                            // relu
        unsigned h = hash_u32((unsigned)i ^ seed);
        v = ((h & 0xFFFFFFu) < keep_thresh) ? v * (1.0f / KEEP_P) : 0.0f;
        Xh[i] = (_Float16)v;
    }
}

__global__ void finalize_out_kernel(const float* __restrict__ outacc,
                                    const float* __restrict__ bias,
                                    float* __restrict__ out, long long N2) {
    for (long long i = (long long)blockIdx.x * blockDim.x + threadIdx.x; i < N2;
         i += (long long)gridDim.x * blockDim.x) {
        long long n = i >> 1; int c = (int)(i & 1);
        out[i] = outacc[n * 64 + c] + bias[c];
    }
}

// ---------------------------------------------------------------------------

struct LayerP { const float *wl, *bl, *wr, *br, *we, *att, *bias; int K, D; };

extern "C" void kernel_launch(void* const* d_in, const int* in_sizes, int n_in,
                              void* d_out, int out_size, void* d_ws, size_t ws_size,
                              hipStream_t stream) {
    const float* x     = (const float*)d_in[0];
    const int*   ei    = (const int*)d_in[1];          // [2, E] int32
    const float* eattr = (const float*)d_in[2];

    const long long N  = in_sizes[0] / IN_DIM;
    const long long E  = in_sizes[2] / EDGE_DIM;
    const long long Ef = E + N;
    const int* src = ei;
    const int* dst = ei + E;

    // params: l1..l3, each (wl, bl, wr, br, we, att, bias) — insertion order
    LayerP L[3];
    for (int l = 0; l < 3; ++l) {
        int b = 3 + 7 * l;
        L[l].wl   = (const float*)d_in[b + 0];
        L[l].bl   = (const float*)d_in[b + 1];
        L[l].wr   = (const float*)d_in[b + 2];
        L[l].br   = (const float*)d_in[b + 3];
        L[l].we   = (const float*)d_in[b + 4];
        L[l].att  = (const float*)d_in[b + 5];
        L[l].bias = (const float*)d_in[b + 6];
        L[l].K = (l == 0) ? IN_DIM : HID;
        L[l].D = (l == 2) ? OUT_DIM : HID;
    }

    // --- workspace carve-out (256B aligned) ---
    char* wsb = (char*)d_ws;
    size_t off = 0;
    auto carve = [&](size_t bytes) -> void* {
        void* p = wsb + off;
        off += (bytes + 255) & ~(size_t)255;
        return p;
    };
    _Float16* Xh       = (_Float16*)carve((size_t)N * 64 * sizeof(_Float16));
    _Float16* WtL      = (_Float16*)carve(64 * 64 * sizeof(_Float16));
    _Float16* WtR      = (_Float16*)carve(64 * 64 * sizeof(_Float16));
    float*    biaspL   = (float*)carve(64 * sizeof(float));
    float*    biaspR   = (float*)carve(64 * sizeof(float));
    float*    xl       = (float*)carve((size_t)N * 64 * sizeof(float));
    float*    xr       = (float*)carve((size_t)N * 64 * sizeof(float));
    float*    outacc   = (float*)carve((size_t)N * 64 * sizeof(float));
    float*    loopattr = (float*)carve((size_t)N * EDGE_DIM * sizeof(float));
    float*    cnt      = (float*)carve((size_t)N * sizeof(float));
    unsigned* mxe      = (unsigned*)carve((size_t)N * sizeof(unsigned));
    float*    den      = (float*)carve((size_t)N * sizeof(float));
    float*    logits   = (float*)carve((size_t)Ef * sizeof(float));
    float*    exv      = (float*)carve((size_t)Ef * sizeof(float));
    (void)ws_size; (void)n_in;

    const int TB = 256;
    auto blocks = [&](long long n) -> unsigned {
        long long b = (n + TB - 1) / TB;
        return (unsigned)(b < 1 ? 1 : b);
    };

    // --- stage 0: self-loop attrs (mean of incoming edge_attr) ---
    fill_f32_kernel<<<blocks(N), TB, 0, stream>>>(cnt, 0.0f, N);
    fill_f32_kernel<<<blocks(N * EDGE_DIM), TB, 0, stream>>>(loopattr, 0.0f, N * EDGE_DIM);
    loop_accum_kernel<<<blocks(E * EDGE_DIM), TB, 0, stream>>>(dst, eattr, loopattr, cnt, E * EDGE_DIM);
    loop_div_kernel<<<blocks(N * EDGE_DIM), TB, 0, stream>>>(loopattr, cnt, N * EDGE_DIM);

    // --- stage 0b: f16 features for layer-1 GEMM ---
    cvt_x_kernel<<<blocks(N * 64), TB, 0, stream>>>(x, IN_DIM, Xh, N * 64);

    const unsigned gemm_grid = (unsigned)((N + 31) / 32);

    for (int l = 0; l < 3; ++l) {
        const LayerP& P = L[l];

        // weight prep + WMMA projections
        prep_wt_kernel<<<16, TB, 0, stream>>>(P.wl, P.K, P.D, P.bl, WtL, biaspL);
        prep_wt_kernel<<<16, TB, 0, stream>>>(P.wr, P.K, P.D, P.br, WtR, biaspR);
        gemm_wmma_kernel<<<gemm_grid, TB, 0, stream>>>(Xh, WtL, biaspL, xl, (int)N);
        gemm_wmma_kernel<<<gemm_grid, TB, 0, stream>>>(Xh, WtR, biaspR, xr, (int)N);

        // per-layer scatter state
        fill_u32_kernel<<<blocks(N), TB, 0, stream>>>(mxe, ENC_NEG_INF, N);
        fill_f32_kernel<<<blocks(N), TB, 0, stream>>>(den, 0.0f, N);
        fill_f32_kernel<<<blocks(N * 64), TB, 0, stream>>>(outacc, 0.0f, N * 64);

        // segment softmax + aggregation
        edge_logits_kernel<<<blocks(Ef), TB, 0, stream>>>(
            src, dst, eattr, loopattr, xl, xr, P.we, P.att, P.D, E, Ef, logits, mxe);
        edge_exp_kernel<<<blocks(Ef), TB, 0, stream>>>(dst, logits, mxe, den, exv, E, Ef);
        edge_scatter_kernel<<<blocks(Ef * 32), TB, 0, stream>>>(
            src, dst, exv, den, xl, outacc, P.D, E, Ef);

        if (l < 2) {
            finalize_hidden_kernel<<<blocks(N * 64), TB, 0, stream>>>(
                outacc, P.bias, 0x9E3779B9u * (unsigned)(l + 1), Xh, N * 64);
        } else {
            finalize_out_kernel<<<blocks(N * OUT_DIM), TB, 0, stream>>>(
                outacc, P.bias, (float*)d_out, N * OUT_DIM);
        }
    }
}
